// MCLLayer_45088566674112
// MI455X (gfx1250) — matrix operand order repaired and checked by reading, not compiled, not run on hardware
//
#include <hip/hip_runtime.h>
#include <hip/hip_bf16.h>
#include <stdint.h>

// ---------------------------------------------------------------------------
// MCL layer for MI455X (gfx1250): WMMA f16 GEMM (2x2 register-blocked) +
// bipartite Katz reduction via 25x25 Schur complement.
//   query_feat:   (16, 75, 640, 5, 5) f32
//   support_feat: (16, 25, 640, 5, 5) f32
//   output:       (1200, 5) f32
// ---------------------------------------------------------------------------

typedef __attribute__((ext_vector_type(16))) _Float16 v16h;
typedef __attribute__((ext_vector_type(8)))  _Float16 v8h;
typedef __attribute__((ext_vector_type(8)))  float    v8f;

#define B_      16
#define Q_      75
#define C_      640
#define HW_     25
#define WAY_    5
#define MS_     625            // way*shot*hw
#define MSPAD_  640            // padded support columns
#define MQ_     25
#define MQPAD_  32
#define GAMMA_   20.0f
#define GAMMA2_  10.0f

union FragAB { v16h v; v8h h[2]; };

static __device__ __forceinline__ v8f wmma_f16(const FragAB& a, const FragAB& b,
                                               v8f c) {
    return __builtin_amdgcn_wmma_f32_16x16x32_f16(false, a.v, false, b.v,
                                                  (short)0, c, false, false);
}

// ---------------------------------------------------------------------------
// K1: normalize one support slab (b, way, shot) = 640x25 f32, write f16 into
//     sf16[b][col][c] (col = way*125 + shot*25 + pos), columns 625..639 stay 0
//     from the preceding memset.
// ---------------------------------------------------------------------------
__global__ void mcl_support_norm(const float* __restrict__ support,
                                 _Float16* __restrict__ sf16) {
    extern __shared__ char smem[];
    float* raw   = (float*)smem;          // 640*25 f32 = 64000 B
    float* scale = (float*)(smem + 64000);

    const int bs   = blockIdx.x;          // 0..399
    const int b    = bs / 25;
    const int s    = bs % 25;             // way*5 + shot
    const int way  = s / 5;
    const int shot = s % 5;
    const int tid  = threadIdx.x;

    const float* in = support + (size_t)(b * 25 + s) * (C_ * HW_);
    for (int idx = tid; idx < C_ * HW_; idx += 256) raw[idx] = in[idx];
    __syncthreads();

    if (tid < HW_) {
        float ss = 0.f;
        for (int c = 0; c < C_; ++c) { float v = raw[c * HW_ + tid]; ss += v * v; }
        float nrm = sqrtf(ss);
        float s1  = 1.0f / fmaxf(nrm, 1e-12f);       // first normalize (EPS_FN)
        float n2  = nrm * s1;                        // norm after first pass
        scale[tid] = s1 / (1e-16f + n2);             // second normalize (EPS_L2)
    }
    __syncthreads();

    _Float16* out_b = sf16 + (size_t)b * MSPAD_ * C_;
    for (int idx = tid; idx < C_ * HW_; idx += 256) {
        int c = idx / HW_, p = idx % HW_;
        int col = way * 125 + shot * 25 + p;
        out_b[(size_t)col * C_ + c] = (_Float16)(raw[idx] * scale[p]);
    }
}

// ---------------------------------------------------------------------------
// K2: one workgroup (256 thr, 8 waves) per n = b*75 + qi.
// ---------------------------------------------------------------------------
__global__ void mcl_main(const float* __restrict__ query,
                         const _Float16* __restrict__ sf16,
                         float* __restrict__ out) {
    extern __shared__ char smem[];
    // Region X: raw query slab (f32, 64000 B) -> reused as S (f16, 32x640)
    float*    rawq = (float*)smem;
    _Float16* Sh   = (_Float16*)smem;
    // Region Y: q-tile (f16, 32x640 = 40960 B) -> reused as Tqs_t (f16, 25x640)
    _Float16* qtile = (_Float16*)(smem + 64000);
    _Float16* Tqs   = (_Float16*)(smem + 64000);
    // Region Z: scalars
    char*  Z      = smem + 64000 + 40960;
    float* rmax   = (float*)Z;            // 32
    float* rsum   = rmax + 32;            // 32
    float* cmax   = rsum + 32;            // 640
    float* csum   = cmax + 640;           // 640
    float* aug    = csum + 640;           // 25 x 26 augmented system
    float* xq     = aug + 650;            // 25
    float* katz   = xq + 25;              // 640
    float* wsum   = katz + 640;           // 8
    float* qscale = wsum + 8;             // 32

    const int n   = blockIdx.x;           // 0..1199
    const int b   = n / Q_;
    const int tid = threadIdx.x;
    const int wv  = tid >> 5;
    const int lane = tid & 31;
    const int lr   = lane & 15;
    const int khalf = (lane >> 4);        // 0: low half, 1: high half

    // ---- Phase 1: async copy of the contiguous 64 KB query slab into LDS ----
    const float* gq = query + (size_t)n * (C_ * HW_);
    for (int it = 0; it < 16; ++it) {
        int v4 = (it * 256 + tid) * 4;    // float index, 16B granules
        if (v4 < C_ * HW_) {
            unsigned ldsoff = (unsigned)(uintptr_t)(rawq + v4);
            const float* ga = gq + v4;
            asm volatile("global_load_async_to_lds_b128 %0, %1, off"
                         :: "v"(ldsoff), "v"(ga) : "memory");
        }
    }
    asm volatile("s_wait_asynccnt 0" ::: "memory");
    __syncthreads();

    // ---- Phase 2: per-row L2 norms (row i = spatial position, vec over c) ----
    if (tid < MQ_) {
        float ss = 0.f;
        for (int c = 0; c < C_; ++c) { float v = rawq[c * HW_ + tid]; ss += v * v; }
        float nrm = sqrtf(ss);
        float s1  = 1.0f / fmaxf(nrm, 1e-12f);
        float n2  = nrm * s1;
        qscale[tid] = s1 / (1e-16f + n2);
    }
    __syncthreads();

    // ---- Phase 3: build f16 A-tile (rows 25..31 zero-padded) ----
    for (int idx = tid; idx < MQPAD_ * C_; idx += 256) {
        int i = idx / C_, c = idx % C_;
        qtile[idx] = (i < MQ_) ? (_Float16)(rawq[c * HW_ + i] * qscale[i])
                               : (_Float16)0.f;
    }
    __syncthreads();

    // ---- Phase 4: S = qf @ sf^T via v_wmma_f32_16x16x32_f16 ----
    // 2x2 register blocking: each wave computes both m-tiles for a pair of
    // n-tiles -> 4 wmma per K-step from 4 ds_load_b128 + 4 global_load_b128
    // (2 operand loads per wmma instead of 4).
    {
        const _Float16* sfb = sf16 + (size_t)b * MSPAD_ * C_;
        const _Float16* ap0 = qtile + (lr)      * C_ + khalf * 8;  // m-tile 0
        const _Float16* ap1 = qtile + (16 + lr) * C_ + khalf * 8;  // m-tile 1
        for (int np = wv; np < 20; np += 8) {                      // n-tile pair
            const int colA = np * 32 + lr;
            const int colB = colA + 16;
            const _Float16* bp0 = sfb + (size_t)colA * C_ + khalf * 8;
            const _Float16* bp1 = sfb + (size_t)colB * C_ + khalf * 8;
            v8f a00 = {}, a01 = {}, a10 = {}, a11 = {};
            for (int k0 = 0; k0 < C_; k0 += 32) {
                FragAB A0, A1, B0, B1;
                A0.h[0] = *(const v8h*)(ap0 + k0);       // K 0..7  (or 8..15)
                A0.h[1] = *(const v8h*)(ap0 + k0 + 16);  // K 16..23 (or 24..31)
                A1.h[0] = *(const v8h*)(ap1 + k0);
                A1.h[1] = *(const v8h*)(ap1 + k0 + 16);
                B0.h[0] = *(const v8h*)(bp0 + k0);       // K 0..7  (or 8..15)
                B0.h[1] = *(const v8h*)(bp0 + k0 + 16);  // K 16..23 (or 24..31)
                B1.h[0] = *(const v8h*)(bp1 + k0);
                B1.h[1] = *(const v8h*)(bp1 + k0 + 16);
                a00 = wmma_f16(A0, B0, a00);
                a01 = wmma_f16(A0, B1, a01);
                a10 = wmma_f16(A1, B0, a10);
                a11 = wmma_f16(A1, B1, a11);
            }
            // C/D layout: lane<16 -> rows mt*16+0..7, lane>=16 -> rows +8..15
            const int r0 = khalf * 8;          // m-tile 0 row base
            const int r1 = 16 + khalf * 8;     // m-tile 1 row base
            const int c0 = np * 32 + lr;       // n-tile 0 column
            const int c1 = c0 + 16;            // n-tile 1 column
#pragma unroll
            for (int r = 0; r < 8; ++r) {
                Sh[(r0 + r) * MSPAD_ + c0] = (_Float16)a00[r];
                Sh[(r0 + r) * MSPAD_ + c1] = (_Float16)a01[r];
                Sh[(r1 + r) * MSPAD_ + c0] = (_Float16)a10[r];
                Sh[(r1 + r) * MSPAD_ + c1] = (_Float16)a11[r];
            }
        }
    }
    __syncthreads();

    // ---- Phase 5: softmax statistics ----
    // rows (gamma=20): wave wv handles rows wv, wv+8, ...
    for (int i = wv; i < MQ_; i += 8) {
        float m = -1e30f;
        for (int j = lane; j < MS_; j += 32) m = fmaxf(m, (float)Sh[i * MSPAD_ + j]);
        for (int msk = 16; msk; msk >>= 1) m = fmaxf(m, __shfl_xor(m, msk, 32));
        float s = 0.f;
        for (int j = lane; j < MS_; j += 32)
            s += __expf(GAMMA_ * ((float)Sh[i * MSPAD_ + j] - m));
        for (int msk = 16; msk; msk >>= 1) s += __shfl_xor(s, msk, 32);
        if (lane == 0) { rmax[i] = m; rsum[i] = s; }
    }
    // columns (gamma=10)
    for (int j = tid; j < MS_; j += 256) {
        float m = -1e30f;
        for (int i = 0; i < MQ_; ++i) m = fmaxf(m, (float)Sh[i * MSPAD_ + j]);
        float s = 0.f;
        for (int i = 0; i < MQ_; ++i)
            s += __expf(GAMMA2_ * ((float)Sh[i * MSPAD_ + j] - m));
        cmax[j] = m; csum[j] = s;
    }
    __syncthreads();

    // ---- Phase 6: T_sq (in place over S) and T_qs^T (into Y region) ----
    for (int idx = tid; idx < MQ_ * MSPAD_; idx += 256) {
        int i = idx / MSPAD_, j = idx - i * MSPAD_;
        float s = (float)Sh[idx];
        float tsq = 0.f, tqs = 0.f;
        if (j < MS_) {
            tsq = __expf(GAMMA_  * (s - rmax[i])) / rsum[i];
            tqs = __expf(GAMMA2_ * (s - cmax[j])) / csum[j];
        }
        Sh[idx]  = (_Float16)tsq;   // T_sq[i][j]
        Tqs[idx] = (_Float16)tqs;   // T_qs^T[i][j] = T_qs[j][i]
    }
    __syncthreads();

    // ---- Phase 7: Schur-complement system (I - 0.25 C B) x = 1 + 0.5 C.1 ----
    if (tid < MQ_) {
        float s = 0.f;
        for (int j = 0; j < MS_; ++j) s += (float)Tqs[tid * MSPAD_ + j];
        aug[tid * 26 + 25] = 1.f + 0.5f * s;
    }
    for (int p = tid; p < MQ_ * MQ_; p += 256) {
        int i = p / MQ_, i2 = p % MQ_;
        float d = 0.f;
        for (int j = 0; j < MS_; ++j)
            d += (float)Tqs[i * MSPAD_ + j] * (float)Sh[i2 * MSPAD_ + j];
        aug[i * 26 + i2] = ((i == i2) ? 1.f : 0.f) - 0.25f * d;
    }

    // ---- Phase 8: 25x25 Gaussian elimination ----
    for (int k = 0; k < MQ_; ++k) {
        __syncthreads();
        if (tid > k && tid < MQ_) {
            float f = aug[tid * 26 + k] / aug[k * 26 + k];
            for (int j = k; j < 26; ++j) aug[tid * 26 + j] -= f * aug[k * 26 + j];
        }
    }
    __syncthreads();
    if (tid == 0) {
        for (int k = MQ_ - 1; k >= 0; --k) {
            float x = aug[k * 26 + 25];
            for (int j = k + 1; j < MQ_; ++j) x -= aug[k * 26 + j] * xq[j];
            xq[k] = x / aug[k * 26 + k];
        }
    }
    __syncthreads();

    // ---- Phase 9: katz_s = 0.5 B x_q ; per-way normalized sums ----
    for (int j = tid; j < MS_; j += 256) {
        float s = 0.f;
        for (int i = 0; i < MQ_; ++i) s += (float)Sh[i * MSPAD_ + j] * xq[i];
        katz[j] = 0.5f * s;
    }
    __syncthreads();
    if (tid < WAY_) {
        float s = 0.f;
        for (int t = 0; t < 125; ++t) s += katz[tid * 125 + t];
        wsum[tid] = s;
    }
    __syncthreads();
    if (tid < WAY_) {
        float tot = wsum[0] + wsum[1] + wsum[2] + wsum[3] + wsum[4];
        out[(size_t)n * WAY_ + tid] = wsum[tid] / tot;
    }
}

// ---------------------------------------------------------------------------
extern "C" void kernel_launch(void* const* d_in, const int* in_sizes, int n_in,
                              void* d_out, int out_size, void* d_ws, size_t ws_size,
                              hipStream_t stream) {
    (void)in_sizes; (void)n_in; (void)out_size; (void)ws_size;
    const float* query   = (const float*)d_in[0];
    const float* support = (const float*)d_in[1];
    float* out           = (float*)d_out;
    _Float16* sf16       = (_Float16*)d_ws;   // [16][640 cols][640 c] f16

    const size_t sf_bytes = (size_t)B_ * MSPAD_ * C_ * sizeof(_Float16); // 13.1 MB
    hipMemsetAsync(d_ws, 0, sf_bytes, stream);

    // K1: 16*25 = 400 slabs, 64 KB + eps dynamic LDS each
    mcl_support_norm<<<dim3(B_ * 25), dim3(256), 64000 + 128, stream>>>(support, sf16);

    // K2: 1200 independent problems
    const size_t smem2 = 64000 + 40960 + 4 * (32 + 32 + 640 + 640 + 650 + 25 + 640 + 8 + 32);
    mcl_main<<<dim3(B_ * Q_), dim3(256), smem2, stream>>>(query, sf16, out);
}
